// HomologicalConnectivityLoss_56573309223820
// MI455X (gfx1250) — compile-verified, matrix operand order checked by reading, not run on hardware
//
#include <hip/hip_runtime.h>
#include <hip/hip_bf16.h>

typedef __attribute__((ext_vector_type(16))) _Float16 v16h;
typedef __attribute__((ext_vector_type(8)))  float    v8f;

#define NN        8192
#define NN_MASK   8191u
#define NN_SHIFT  13
// total elements = 8192*8192 = 67108864
// scan grid: 1024 blocks * 256 threads = 8192 waves (wave32)
// per-wave elements = 67108864 / 8192 = 8192 = 16 WMMA tiles of 512

__global__ void hcl_scan_kernel(const float* __restrict__ adj,
                                unsigned* __restrict__ edgeCnt,
                                unsigned* __restrict__ posAcc,
                                unsigned* __restrict__ edges,
                                unsigned cap)
{
    const int lane = threadIdx.x & 31;
    const unsigned wave = (blockIdx.x * blockDim.x + threadIdx.x) >> 5;
    const unsigned waveBase = wave * 8192u;

    v16h ones;
    #pragma unroll
    for (int k = 0; k < 16; ++k) ones[k] = (_Float16)1.0f;

    v8f c = {};   // f32 accumulator, chained across tiles

    #pragma unroll 1
    for (int it = 0; it < 16; ++it) {
        const unsigned base = waveBase + (unsigned)it * 512u + (unsigned)lane * 16u;
        const float4* p = (const float4*)(adj + base);
        float4 f0 = p[0];
        float4 f1 = p[1];
        float4 f2 = p[2];
        float4 f3 = p[3];
        float v[16] = { f0.x, f0.y, f0.z, f0.w,
                        f1.x, f1.y, f1.z, f1.w,
                        f2.x, f2.y, f2.z, f2.w,
                        f3.x, f3.y, f3.z, f3.w };

        v16h a;
        unsigned mask = 0u;
        #pragma unroll
        for (int k = 0; k < 16; ++k) {
            bool pos = v[k] > 0.0f;
            a[k] = pos ? (_Float16)1.0f : (_Float16)0.0f;
            mask |= pos ? (1u << k) : 0u;
        }

        // count positives on the matrix pipe: D = A * ones + C
        // (row sums replicated across 16 columns -> wave total = sum/16)
        c = __builtin_amdgcn_wmma_f32_16x16x32_f16(
                /*neg_a=*/false, a, /*neg_b=*/false, ones,
                /*c_mod=*/(short)0, c, /*reuse_a=*/false, /*reuse_b=*/false);

        // compact upper-triangle edges (sparse: ~1.2 positives per 512 elems)
        while (mask) {
            int k = __ffs(mask) - 1;
            mask &= mask - 1u;
            unsigned idx = base + (unsigned)k;
            unsigned i = idx >> NN_SHIFT;
            unsigned j = idx & NN_MASK;
            if (j > i) {
                unsigned slot = atomicAdd(edgeCnt, 1u);
                if (slot < cap) edges[slot] = (i << NN_SHIFT) | j;
            }
        }
    }

    // reduce accumulator: each value is an exact small integer in f32
    float s = c[0] + c[1] + c[2] + c[3] + c[4] + c[5] + c[6] + c[7];
    #pragma unroll
    for (int off = 16; off > 0; off >>= 1)
        s += __shfl_down(s, off, 32);
    if (lane == 0) {
        // lane0 holds 16 * (wave positive count); exact division
        atomicAdd(posAcc, (unsigned)(s * 0.0625f + 0.5f));
    }
}

__global__ void hcl_cc_kernel(const unsigned* __restrict__ edgeCnt,
                              const unsigned* __restrict__ posAcc,
                              const unsigned* __restrict__ edges,
                              unsigned cap,
                              float* __restrict__ out)
{
    __shared__ int labels[NN];           // 32 KB of the WGP's 320 KB LDS
    __shared__ int changed;
    __shared__ unsigned ncomp;
    const int tid = threadIdx.x;
    const int nthreads = blockDim.x;

    for (int i = tid; i < NN; i += nthreads) labels[i] = i;
    if (tid == 0) ncomp = 0u;
    __syncthreads();

    unsigned E = *edgeCnt;
    if (E > cap) E = cap;

    // min-label relaxation to fixed point (== reference's while_loop fixpoint)
    while (true) {
        if (tid == 0) changed = 0;
        __syncthreads();
        for (unsigned e = (unsigned)tid; e < E; e += (unsigned)nthreads) {
            unsigned p = edges[e];
            int i = (int)(p >> NN_SHIFT);
            int j = (int)(p & NN_MASK);
            int li = labels[i];
            int lj = labels[j];
            if (lj < li) {
                if (atomicMin(&labels[i], lj) > lj) changed = 1;
            } else if (li < lj) {
                if (atomicMin(&labels[j], li) > li) changed = 1;
            }
        }
        __syncthreads();
        if (!changed) break;
        // pointer jumping accelerates propagation; monotone, label stays in-component
        for (int i = tid; i < NN; i += nthreads) {
            int l = labels[i];
            int m = labels[l];
            if (m < l) atomicMin(&labels[i], m);
        }
        __syncthreads();
    }

    // one node per component keeps its own (minimal) index
    unsigned local = 0u;
    for (int i = tid; i < NN; i += nthreads)
        local += (labels[i] == i) ? 1u : 0u;
    atomicAdd(&ncomp, local);
    __syncthreads();

    if (tid == 0) {
        float nc = (float)ncomp;
        unsigned ne_u = (*posAcc) / 2u;          // includes diagonal, floor-div like reference
        float ne = (float)ne_u;
        float betti = ne - 8192.0f + nc;
        float cyc = fmaxf(0.0f, betti);
        float d = nc - 1.0f;
        out[0] = d * d + cyc * cyc;
    }
}

extern "C" void kernel_launch(void* const* d_in, const int* in_sizes, int n_in,
                              void* d_out, int out_size, void* d_ws, size_t ws_size,
                              hipStream_t stream) {
    (void)in_sizes; (void)n_in; (void)out_size;
    const float* adj = (const float*)d_in[0];
    float* out = (float*)d_out;

    unsigned* ws      = (unsigned*)d_ws;
    unsigned* edgeCnt = ws + 0;
    unsigned* posAcc  = ws + 1;
    unsigned* edges   = ws + 8;
    unsigned cap = (ws_size / 4 > 16) ? (unsigned)(ws_size / 4 - 16) : 0u;

    hipMemsetAsync(d_ws, 0, 32, stream);  // zero counters (capture-safe)

    hcl_scan_kernel<<<1024, 256, 0, stream>>>(adj, edgeCnt, posAcc, edges, cap);
    hcl_cc_kernel<<<1, 1024, 0, stream>>>(edgeCnt, posAcc, edges, cap, out);
}
